// EMAVectorQuantizer_15908558865422
// MI455X (gfx1250) — matrix operand order repaired and checked by reading, not compiled, not run on hardware
//
#include <hip/hip_runtime.h>
#include <hip/hip_bf16.h>

// ---------------------------------------------------------------------------
// VQ quantizer forward for MI455X (gfx1250, wave32, WMMA).
// dist(n,k) = const(n) + [6x^2 | -4x^3 | -4x] . [e^2 | e | e^3] + sum(e^4)
// -> single bf16 GEMM [65536 x 768] x [768 x 1024], f32 accumulate, argmin.
// Register-blocked: each wave computes 16 pixels x 64 codes (4 accumulators
// sharing one A fragment). B tiles staged via async global->LDS when the
// toolchain exposes the gfx1250 builtin (ASYNCcnt path), else VGPR copies.
// ---------------------------------------------------------------------------

typedef __attribute__((ext_vector_type(16))) __bf16 v16bf;
typedef __attribute__((ext_vector_type(8)))  __bf16 v8bf;
typedef __attribute__((ext_vector_type(8)))  float  v8f;
typedef __attribute__((ext_vector_type(4)))  int    v4i;

typedef __attribute__((address_space(1))) v4i* gv4i_ptr;
typedef __attribute__((address_space(3))) v4i* lv4i_ptr;

#define NB     16
#define CH     256
#define HW     4096            // 64*64
#define NPIX   (NB * HW)       // 65536
#define KCODE  1024
#define DEXT   768             // 3 * CH
#define ASTR   776             // padded LDS row stride (halves) to spread banks
#define PIXT   64              // pixels per block
#define KTT    128             // codes per K-tile (B tile rows in LDS)
#define NKT    (KCODE / KTT)   // 8 tiles

#if defined(__has_builtin)
#if __has_builtin(__builtin_amdgcn_global_load_async_to_lds_b128)
#define USE_ASYNC_LDS 1
#endif
#endif

// ---------------------------------------------------------------------------
// Kernel 1: build B matrix [k][e^2 | e | e^3] (bf16) and sum(e^4) (f32).
// ---------------------------------------------------------------------------
__global__ void vq_prep_embed(const float* __restrict__ embed,
                              __bf16* __restrict__ Bmat,
                              float* __restrict__ e4sum) {
    const int k = blockIdx.x;
    const int c = threadIdx.x;               // 0..255
    const float e  = embed[(size_t)k * CH + c];
    const float e2 = e * e;
    __bf16* row = Bmat + (size_t)k * DEXT;
    row[c]            = (__bf16)e2;          // pairs with 6*x^2
    row[CH + c]       = (__bf16)e;           // pairs with -4*x^3
    row[2 * CH + c]   = (__bf16)(e2 * e);    // pairs with -4*x
    float e4 = e2 * e2;
    #pragma unroll
    for (int m = 16; m >= 1; m >>= 1) e4 += __shfl_xor(e4, m, 32);
    __shared__ float wsum[8];
    if ((c & 31) == 0) wsum[c >> 5] = e4;
    __syncthreads();
    if (c == 0) {
        float s = 0.f;
        #pragma unroll
        for (int i = 0; i < 8; ++i) s += wsum[i];
        e4sum[k] = s;
    }
}

// ---------------------------------------------------------------------------
// Kernel 2: distances via WMMA + argmin. One block = 64 pixels, all 1024 codes.
// 8 waves: wave (wp, wk) owns a 16-pixel x 64-code strip: 4 accumulators that
// share one A fragment per k-step (2 A + 8 B ds-loads per 4 WMMAs).
// ---------------------------------------------------------------------------
__global__ void vq_dist_argmin(const float* __restrict__ inp,
                               const __bf16* __restrict__ Bmat,
                               const float* __restrict__ e4sum,
                               int* __restrict__ wsIdx,
                               float* __restrict__ outIdx) {
    extern __shared__ char smem[];
    __bf16* ldsA = (__bf16*)smem;                          // 64 x 776 halves
    __bf16* ldsB = (__bf16*)(smem + PIXT * ASTR * 2);      // 128 x 776 halves
    float*  minv = (float*)(smem + (PIXT + KTT) * ASTR * 2);         // [2][64]
    int*    mini = (int*)(smem + (PIXT + KTT) * ASTR * 2 + 2 * PIXT * 4);

    const int t    = threadIdx.x;
    const int lane = t & 31;
    const int wave = t >> 5;            // 0..7
    const int wp   = wave & 3;          // pixel-tile quadrant
    const int wk   = wave >> 2;         // 64-code half of the 128-wide K tile

    const int n0  = blockIdx.x * PIXT;
    const int bb  = n0 / HW;
    const int hw0 = n0 - bb * HW;

    // ---- stage A tile: x from [B,C,HW] layout, coalesced along HW ----------
    {
        const int hwo = t & 63;
        const float* base = inp + (size_t)bb * CH * HW + hw0 + hwo;
        for (int c = (t >> 6); c < CH; c += 4) {
            const float x  = base[(size_t)c * HW];
            const float x2 = x * x;
            __bf16* row = ldsA + (size_t)hwo * ASTR;
            row[c]          = (__bf16)(6.0f * x2);
            row[CH + c]     = (__bf16)(-4.0f * x2 * x);
            row[2 * CH + c] = (__bf16)(-4.0f * x);
        }
    }

    // fragment base addresses (ISA 7.12.2 layouts, wave32)
    const __bf16* aBase = ldsA + (size_t)(wp * 16 + (lane & 15)) * ASTR
                               + ((lane & 16) ? 8 : 0);
    const __bf16* bBase0 = ldsB + (size_t)(wk * 64 + (lane & 15)) * ASTR
                                + ((lane & 16) ? 8 : 0);
    const __bf16* bBase[4];
    #pragma unroll
    for (int sub = 0; sub < 4; ++sub)
        bBase[sub] = bBase0 + (size_t)(sub * 16) * ASTR;

    float runmin[8];
    int   runidx[8];
    #pragma unroll
    for (int j = 0; j < 8; ++j) { runmin[j] = 3.4e38f; runidx[j] = 0; }

    for (int kt = 0; kt < NKT; ++kt) {
        // ---- B-tile staging: 128 rows x 768 halves -------------------------
        const __bf16* gB = Bmat + (size_t)(kt * KTT) * DEXT;
#if defined(USE_ASYNC_LDS)
        // gfx1250 async global->LDS: no VGPR round-trip, tracked by ASYNCcnt.
        for (int idx = t; idx < KTT * (DEXT / 8); idx += 256) {
            const int kk = idx / (DEXT / 8);
            const int dd = (idx - kk * (DEXT / 8)) * 8;
            __builtin_amdgcn_global_load_async_to_lds_b128(
                (gv4i_ptr)(gB + (size_t)kk * DEXT + dd),
                (lv4i_ptr)(ldsB + (size_t)kk * ASTR + dd),
                0, 0);
        }
#if __has_builtin(__builtin_amdgcn_s_wait_asynccnt)
        __builtin_amdgcn_s_wait_asynccnt(0);
#else
        asm volatile("s_wait_asynccnt 0x0" ::: "memory");
#endif
#else
        for (int idx = t; idx < KTT * (DEXT / 8); idx += 256) {
            const int kk = idx / (DEXT / 8);
            const int dd = (idx - kk * (DEXT / 8)) * 8;
            *(uint4*)(ldsB + (size_t)kk * ASTR + dd) =
                *(const uint4*)(gB + (size_t)kk * DEXT + dd);
        }
#endif
        if (kt + 1 < NKT)
            __builtin_prefetch(Bmat + (size_t)((kt + 1) * KTT) * DEXT, 0, 1);
        __syncthreads();

        // ---- 16x64 strip: 24 k-steps x 4 WMMAs, A fragment shared ----------
        v8f acc[4];
        #pragma unroll
        for (int sub = 0; sub < 4; ++sub) acc[sub] = (v8f){};
        #pragma unroll
        for (int dc = 0; dc < DEXT / 32; ++dc) {
            const int d0 = dc * 32;
            v8bf alo = *(const v8bf*)(aBase + d0);
            v8bf ahi = *(const v8bf*)(aBase + d0 + 16);
            v16bf a = __builtin_shufflevector(alo, ahi, 0,1,2,3,4,5,6,7,
                                                         8,9,10,11,12,13,14,15);
            #pragma unroll
            for (int sub = 0; sub < 4; ++sub) {
                v8bf blo = *(const v8bf*)(bBase[sub] + d0);
                v8bf bhi = *(const v8bf*)(bBase[sub] + d0 + 16);
                v16bf b = __builtin_shufflevector(blo, bhi, 0,1,2,3,4,5,6,7,
                                                             8,9,10,11,12,13,14,15);
                acc[sub] = __builtin_amdgcn_wmma_f32_16x16x32_bf16(
                               false, a, false, b, (short)0, acc[sub], false, false);
            }
        }

        // ---- fold in sum(e^4) and track running min ------------------------
        #pragma unroll
        for (int sub = 0; sub < 4; ++sub) {
            const int kglob = kt * KTT + wk * 64 + sub * 16 + (lane & 15);
            const float e4v = e4sum[kglob];
            #pragma unroll
            for (int j = 0; j < 8; ++j) {
                const float s = acc[sub][j] + e4v;
                if (s < runmin[j]) { runmin[j] = s; runidx[j] = kglob; }
            }
        }
        __syncthreads();   // before next tile overwrites ldsB
    }

    // ---- reduce across the 16 lanes holding the same pixel rows ------------
    #pragma unroll
    for (int j = 0; j < 8; ++j) {
        #pragma unroll
        for (int m = 1; m <= 8; m <<= 1) {
            const float ov = __shfl_xor(runmin[j], m, 32);
            const int   oi = __shfl_xor(runidx[j], m, 32);
            if (ov < runmin[j] || (ov == runmin[j] && oi < runidx[j])) {
                runmin[j] = ov; runidx[j] = oi;
            }
        }
    }
    if ((lane & 15) == 0) {
        const int half = lane >> 4;     // VGPR j -> row j (lanes<16) or 8+j
        #pragma unroll
        for (int j = 0; j < 8; ++j) {
            const int p = wp * 16 + half * 8 + j;
            minv[wk * PIXT + p] = runmin[j];
            mini[wk * PIXT + p] = runidx[j];
        }
    }
    __syncthreads();

    // ---- combine the two k-halves, emit indices ----------------------------
    if (t < PIXT) {
        const float v0 = minv[t], v1 = minv[PIXT + t];
        const int   i0 = mini[t], i1 = mini[PIXT + t];
        const int kbest = (v1 < v0 || (v1 == v0 && i1 < i0)) ? i1 : i0;
        wsIdx[n0 + t]  = kbest;
        outIdx[n0 + t] = (float)kbest;
    }
}

// ---------------------------------------------------------------------------
// Kernel 3: gather embed rows -> [B,C,H,W], straight-through value, MSE loss.
// ---------------------------------------------------------------------------
__global__ void vq_quantize_loss(const float* __restrict__ inp,
                                 const float* __restrict__ embed,
                                 const int* __restrict__ wsIdx,
                                 float* __restrict__ outQ,
                                 float* __restrict__ outLoss) {
    const int bid = blockIdx.x;              // 65536 = 16(b)*256(c)*16(hw chunk)
    const int hwChunk = bid & 15;
    const int c = (bid >> 4) & 255;
    const int b = bid >> 12;
    const int hw = hwChunk * 256 + threadIdx.x;

    const int idx = wsIdx[b * HW + hw];
    const size_t off = (size_t)b * CH * HW + (size_t)c * HW + hw;
    const float xin = inp[off];
    const float q   = embed[(size_t)idx * CH + c];
    outQ[off] = xin + (q - xin);             // straight-through value == q
    const float d = q - xin;
    float val = d * d;
    #pragma unroll
    for (int m = 16; m >= 1; m >>= 1) val += __shfl_xor(val, m, 32);
    __shared__ float wsum[8];
    if ((threadIdx.x & 31) == 0) wsum[threadIdx.x >> 5] = val;
    __syncthreads();
    if (threadIdx.x == 0) {
        float s = 0.f;
        #pragma unroll
        for (int i = 0; i < 8; ++i) s += wsum[i];
        // loss = 0.25 * mean over 16 * 256 * 64 * 64 elements
        atomicAdd(outLoss, s * (0.25f / (float)((size_t)NB * CH * HW)));
    }
}

// ---------------------------------------------------------------------------
extern "C" void kernel_launch(void* const* d_in, const int* in_sizes, int n_in,
                              void* d_out, int out_size, void* d_ws, size_t ws_size,
                              hipStream_t stream) {
    const float* inp   = (const float*)d_in[0];   // [16,256,64,64]
    const float* embed = (const float*)d_in[1];   // [1024,256]

    // workspace layout
    __bf16* Bmat  = (__bf16*)d_ws;                               // 1.5 MB
    float*  e4sum = (float*)((char*)d_ws + (size_t)KCODE * DEXT * 2);
    int*    wsIdx = (int*)((char*)d_ws + (size_t)KCODE * DEXT * 2 + KCODE * 4);

    // output layout: quantized_st | indices(as float) | loss
    float* outQ    = (float*)d_out;
    float* outIdx  = outQ + (size_t)NB * CH * HW;
    float* outLoss = outIdx + NPIX;

    const size_t smemBytes = (size_t)(PIXT + KTT) * ASTR * 2   // A + B tiles
                           + 2 * PIXT * 4 + 2 * PIXT * 4;      // minv + mini
    (void)hipFuncSetAttribute((const void*)vq_dist_argmin,
                              hipFuncAttributeMaxDynamicSharedMemorySize,
                              (int)smemBytes);

    vq_prep_embed<<<KCODE, CH, 0, stream>>>(embed, Bmat, e4sum);
    vq_dist_argmin<<<NPIX / PIXT, 256, smemBytes, stream>>>(
        inp, Bmat, e4sum, wsIdx, outIdx);
    (void)hipMemsetAsync(outLoss, 0, sizeof(float), stream);
    vq_quantize_loss<<<(size_t)NB * CH * (HW / 256), 256, 0, stream>>>(
        inp, embed, wsIdx, outQ, outLoss);
}
